// OptionEncoder_Recurrent_76665166234149
// MI455X (gfx1250) — matrix operand order repaired and checked by reading, not compiled, not run on hardware
//
#include <hip/hip_runtime.h>

// Problem constants (from the reference)
#define BATCH 512
#define TT    1024
#define DD    64          // input_size
#define HH    32          // hidden size
#define CC    4           // categorical dim (L=1)
#define IN0   68          // D + L*C
#define GATES 128         // 4*H
#define K0    128         // layer-0 fused K: [x(64)|ct(4)|h1(32)|pad(28)]
#define K1    64          // layer-1 fused K: [h1(32)|h2(32)]
#define MTILE 16          // WMMA M
#define NTILES 8          // 128 gate cols / 16
#define KS0   4           // K0/32
#define KS1   2           // K1/32

#if defined(__HIP_DEVICE_COMPILE__)
typedef _Float16 v16h __attribute__((ext_vector_type(16)));
typedef _Float16 v8h  __attribute__((ext_vector_type(8)));
typedef float    v8f  __attribute__((ext_vector_type(8)));
union AFrag { v16h v; v8h h[2]; };
union Acc   { v8f  v; float f[8]; };

// Branch-free transcendentals: single v_exp_f32 / v_rcp_f32, no exec-mask
// branches on the recurrence critical path.
__device__ __forceinline__ float fast_sigmoid(float x) {
  return __builtin_amdgcn_rcpf(1.0f + __expf(-x));
}
__device__ __forceinline__ float fast_tanh(float x) {
  float e = __expf(-2.0f * x);
  return (1.0f - e) * __builtin_amdgcn_rcpf(1.0f + e);
}
#endif

__global__ __launch_bounds__(256, 1)
void option_encoder_recurrent(
    const float* __restrict__ states, const float* __restrict__ ptau,
    const float* __restrict__ gumbel,
    const float* __restrict__ w_ih0, const float* __restrict__ w_hh0,
    const float* __restrict__ b_ih0, const float* __restrict__ b_hh0,
    const float* __restrict__ w_ih1, const float* __restrict__ w_hh1,
    const float* __restrict__ b_ih1, const float* __restrict__ b_hh1,
    const float* __restrict__ w_lin, const float* __restrict__ b_lin,
    float* __restrict__ out)
{
#if defined(__HIP_DEVICE_COMPILE__)
  // -------- LDS: weights pre-swizzled into WMMA B-fragment order ----------
  // B 32x16 f16 fragment order used here (wave32, contiguous k, same as A): lane l holds column n = l&15,
  // k = (l>>4)*16 + slot, 16 f16 contiguous per lane -> 2x ds_load_b128.
  __shared__ alignas(16) _Float16 wf0[NTILES * KS0 * 512];  // 32 KB
  __shared__ alignas(16) _Float16 wf1[NTILES * KS1 * 512];  // 16 KB
  __shared__ alignas(16) _Float16 xbuf[MTILE][K0];          // layer-0 A operand
  __shared__ alignas(16) _Float16 abuf[MTILE][K1];          // layer-1 A operand
  __shared__ float zbuf[MTILE][GATES];                      // gate staging
  __shared__ float sWlin[CC][HH];
  __shared__ float sBlin[CC];
  __shared__ float sTau;

  const int tid  = threadIdx.x;
  const int lane = tid & 31;
  const int wid  = tid >> 5;              // 8 waves: one gate N-tile each
  const int b0   = blockIdx.x * MTILE;

  // ---- stage fused layer-0 weights W0[K0=128][N=128] as fragments
  for (int e = tid; e < NTILES * KS0 * 512; e += 256) {
    int nt = e >> 11, ks = (e >> 9) & 3, ln = (e >> 4) & 31, slot = e & 15;
    int n = nt * 16 + (ln & 15);
    int k = ks * 32 + (ln >> 4) * 16 + slot;
    float v = 0.f;
    if (k < IN0)           v = w_ih0[n * IN0 + k];          // x_t and ct rows
    else if (k < IN0 + HH) v = w_hh0[n * HH + (k - IN0)];   // h1 rows
    wf0[e] = (_Float16)v;
  }
  // ---- stage fused layer-1 weights W1[K1=64][N=128]
  for (int e = tid; e < NTILES * KS1 * 512; e += 256) {
    int nt = e >> 10, ks = (e >> 9) & 1, ln = (e >> 4) & 31, slot = e & 15;
    int n = nt * 16 + (ln & 15);
    int k = ks * 32 + (ln >> 4) * 16 + slot;
    float v = (k < HH) ? w_ih1[n * HH + k] : w_hh1[n * HH + (k - HH)];
    wf1[e] = (_Float16)v;
  }
  if (tid < CC * HH) sWlin[tid >> 5][tid & 31] = w_lin[tid];
  if (tid < CC)      sBlin[tid] = b_lin[tid];
  if (tid == 0)      sTau = ptau[0];

  // ---- activations init: h1=h2=0, ct = one-hot(0)
  for (int e = tid; e < MTILE * K0; e += 256) {
    int k = e & (K0 - 1);
    ((_Float16*)xbuf)[e] = (k == DD) ? (_Float16)1.0f : (_Float16)0.0f;
  }
  for (int e = tid; e < MTILE * K1; e += 256)
    ((_Float16*)abuf)[e] = (_Float16)0.0f;

  // ---- per-thread cell state + fused biases for units (em, jj), (em, jj+1)
  const int em = tid >> 4;
  const int jj = (tid & 15) * 2;
  float c1[2] = {0.f, 0.f}, c2[2] = {0.f, 0.f};
  float bz0[4][2], bz1[4][2];
  #pragma unroll
  for (int g = 0; g < 4; ++g)
    #pragma unroll
    for (int u = 0; u < 2; ++u) {
      bz0[g][u] = b_ih0[g * HH + jj + u] + b_hh0[g * HH + jj + u];
      bz1[g][u] = b_ih1[g * HH + jj + u] + b_hh1[g * HH + jj + u];
    }

  __syncthreads();
  const float tau_inv = __builtin_amdgcn_rcpf(sTau);

  for (int t = 0; t < TT; ++t) {
    // ========== phase A: stage x_t (16 x 64 f32 -> f16), prefetch t+1 =====
    {
      const int m = tid >> 4;
      const int k = (tid & 15) * 4;
      const float* src = states + (size_t)(b0 + m) * TT * DD + (size_t)t * DD + k;
      float4 xv = *(const float4*)src;
      xbuf[m][k + 0] = (_Float16)xv.x;  xbuf[m][k + 1] = (_Float16)xv.y;
      xbuf[m][k + 2] = (_Float16)xv.z;  xbuf[m][k + 3] = (_Float16)xv.w;
      if (t + 1 < TT) __builtin_prefetch(src + DD, 0, 3);   // next timestep
    }
    __syncthreads();

    // ========== phase B: z0[16x128] = xbuf @ W0 (f16 WMMA, f32 acc) =======
    {
      Acc acc;
      #pragma unroll
      for (int i = 0; i < 8; ++i) acc.f[i] = 0.f;
      const int m = lane & 15, kh = lane >> 4;
      #pragma unroll
      for (int ks = 0; ks < KS0; ++ks) {
        AFrag a, b;
        const _Float16* ap = &xbuf[m][ks * 32 + kh * 16];   // A 16x32 layout
        a.h[0] = *(const v8h*)ap;
        a.h[1] = *(const v8h*)(ap + 8);
        const _Float16* bp = &wf0[(wid * KS0 + ks) * 512 + lane * 16];
        b.h[0] = *(const v8h*)bp;
        b.h[1] = *(const v8h*)(bp + 8);
        acc.v = __builtin_amdgcn_wmma_f32_16x16x32_f16(
            false, a.v, false, b.v, (short)0, acc.v, false, false);
      }
      const int n  = wid * 16 + (lane & 15);
      const int mb = (lane >> 4) * 8;
      #pragma unroll
      for (int r = 0; r < 8; ++r) zbuf[mb + r][n] = acc.f[r];  // C/D layout
    }
    __syncthreads();

    // ========== phase C: layer-0 LSTM cell (2 units / thread) =============
    #pragma unroll
    for (int u = 0; u < 2; ++u) {
      const int j = jj + u;
      float zi = zbuf[em][j]          + bz0[0][u];
      float zf = zbuf[em][HH + j]     + bz0[1][u];
      float zg = zbuf[em][2 * HH + j] + bz0[2][u];
      float zo = zbuf[em][3 * HH + j] + bz0[3][u];
      c1[u] = fast_sigmoid(zf) * c1[u] + fast_sigmoid(zi) * fast_tanh(zg);
      _Float16 h1h = (_Float16)(fast_sigmoid(zo) * fast_tanh(c1[u]));
      xbuf[em][DD + CC + j] = h1h;   // feeds next step's layer-0 input
      abuf[em][j]           = h1h;   // feeds layer-1 now
    }
    __syncthreads();

    // ========== phase D: z1[16x128] = [h1|h2] @ W1 ========================
    {
      Acc acc;
      #pragma unroll
      for (int i = 0; i < 8; ++i) acc.f[i] = 0.f;
      const int m = lane & 15, kh = lane >> 4;
      #pragma unroll
      for (int ks = 0; ks < KS1; ++ks) {
        AFrag a, b;
        const _Float16* ap = &abuf[m][ks * 32 + kh * 16];
        a.h[0] = *(const v8h*)ap;
        a.h[1] = *(const v8h*)(ap + 8);
        const _Float16* bp = &wf1[(wid * KS1 + ks) * 512 + lane * 16];
        b.h[0] = *(const v8h*)bp;
        b.h[1] = *(const v8h*)(bp + 8);
        acc.v = __builtin_amdgcn_wmma_f32_16x16x32_f16(
            false, a.v, false, b.v, (short)0, acc.v, false, false);
      }
      const int n  = wid * 16 + (lane & 15);
      const int mb = (lane >> 4) * 8;
      #pragma unroll
      for (int r = 0; r < 8; ++r) zbuf[mb + r][n] = acc.f[r];
    }
    __syncthreads();

    // ========== phase E: layer-1 LSTM cell ================================
    #pragma unroll
    for (int u = 0; u < 2; ++u) {
      const int j = jj + u;
      float zi = zbuf[em][j]          + bz1[0][u];
      float zf = zbuf[em][HH + j]     + bz1[1][u];
      float zg = zbuf[em][2 * HH + j] + bz1[2][u];
      float zo = zbuf[em][3 * HH + j] + bz1[3][u];
      c2[u] = fast_sigmoid(zf) * c2[u] + fast_sigmoid(zi) * fast_tanh(zg);
      abuf[em][HH + j] = (_Float16)(fast_sigmoid(zo) * fast_tanh(c2[u]));  // h2
    }
    __syncthreads();

    // ========== phase F: linear (32->4) + Gumbel-softmax, write ct/out ====
    if (tid < MTILE) {
      const int m = tid, b = b0 + m;
      float h2v[HH];
      #pragma unroll
      for (int j = 0; j < HH; ++j) h2v[j] = (float)abuf[m][HH + j];
      const float* gv = gumbel + ((size_t)t * BATCH + b) * CC;
      float logit[CC], mx = -1e30f;
      #pragma unroll
      for (int c = 0; c < CC; ++c) {
        float q = sBlin[c];
        #pragma unroll
        for (int j = 0; j < HH; ++j) q += h2v[j] * sWlin[c][j];
        logit[c] = (q + gv[c]) * tau_inv;
        mx = fmaxf(mx, logit[c]);
      }
      float ex[CC], s = 0.f;
      #pragma unroll
      for (int c = 0; c < CC; ++c) { ex[c] = __expf(logit[c] - mx); s += ex[c]; }
      const float inv = __builtin_amdgcn_rcpf(s);
      float* op = out + ((size_t)b * TT + t) * CC;
      #pragma unroll
      for (int c = 0; c < CC; ++c) {
        float y = ex[c] * inv;
        op[c] = y;                          // output
        xbuf[m][DD + c] = (_Float16)y;      // ct feedback into layer-0 input
      }
      if (t + 1 < TT) __builtin_prefetch(gv + BATCH * CC, 0, 3);
    }
    __syncthreads();
  }
#endif
}

extern "C" void kernel_launch(void* const* d_in, const int* in_sizes, int n_in,
                              void* d_out, int out_size, void* d_ws, size_t ws_size,
                              hipStream_t stream) {
  (void)in_sizes; (void)n_in; (void)d_ws; (void)ws_size; (void)out_size;
  const float* states = (const float*)d_in[0];
  const float* tau    = (const float*)d_in[1];
  const float* gumbel = (const float*)d_in[2];
  const float* w_ih0  = (const float*)d_in[3];
  const float* w_hh0  = (const float*)d_in[4];
  const float* b_ih0  = (const float*)d_in[5];
  const float* b_hh0  = (const float*)d_in[6];
  const float* w_ih1  = (const float*)d_in[7];
  const float* w_hh1  = (const float*)d_in[8];
  const float* b_ih1  = (const float*)d_in[9];
  const float* b_hh1  = (const float*)d_in[10];
  const float* w_lin  = (const float*)d_in[11];
  const float* b_lin  = (const float*)d_in[12];
  float* outp = (float*)d_out;

  dim3 grid(BATCH / MTILE);   // 32 persistent workgroups, one 16-row chain each
  dim3 block(256);            // 8 wave32 -> one gate N-tile per wave
  hipLaunchKernelGGL(option_encoder_recurrent, grid, block, 0, stream,
                     states, tau, gumbel, w_ih0, w_hh0, b_ih0, b_hh0,
                     w_ih1, w_hh1, b_ih1, b_hh1, w_lin, b_lin, outp);
}
